// Mamba2Layer_14834817040751
// MI455X (gfx1250) — matrix operand order repaired and checked by reading, not compiled, not run on hardware
//
#include <hip/hip_runtime.h>
#include <hip/hip_bf16.h>
#include <math.h>

// ---------------- Types ----------------
typedef __attribute__((ext_vector_type(16))) _Float16 v16h;
typedef __attribute__((ext_vector_type(8)))  _Float16 v8h_t;
typedef __attribute__((ext_vector_type(8)))  float    v8f;

static constexpr int Bn = 2;
static constexpr int Tn = 1024;
static constexpr int Hn = 16;
static constexpr int Dn = 1024;
static constexpr int CH = 64;   // channels per head
static constexpr int SN = 64;   // state N
static constexpr int PR = 256;  // fused projection width: 192 (XBC) + 64 (gate)

// ---------------- WMMA helpers ----------------
__device__ __forceinline__ v8f wmma_f16(v16h a, v16h b, v8f c) {
  // (neg_a, A, neg_b, B, c_mod, C, reuse_a, reuse_b)
  return __builtin_amdgcn_wmma_f32_16x16x32_f16(false, a, false, b, (short)0, c,
                                                false, false);
}

__device__ __forceinline__ v8f zero8() {
  v8f z = {0.f, 0.f, 0.f, 0.f, 0.f, 0.f, 0.f, 0.f};
  return z;
}

// A fragment (16x32 f16, MxK): lane row = lane&15, hs = lane>>4.
// element e: K = (e>>3)*16 + hs*8 + (e&7). rowk points at row start (+kbase).
__device__ __forceinline__ v16h load_a_f16v(const _Float16* rowk, int hs) {
  v8h_t lo = *(const v8h_t*)(rowk + hs * 8);
  v8h_t hi = *(const v8h_t*)(rowk + 16 + hs * 8);
  v16h a;
#pragma unroll
  for (int i = 0; i < 8; ++i) {
    a[i]     = lo[i];
    a[i + 8] = hi[i];
  }
  return a;
}

// scalar variant (LDS rows with padded stride -> not 16B aligned)
__device__ __forceinline__ v16h load_a_f16s(const _Float16* rowk, int hs) {
  v16h a;
  const _Float16* p0 = rowk + hs * 8;
  const _Float16* p1 = rowk + 16 + hs * 8;
#pragma unroll
  for (int i = 0; i < 8; ++i) {
    a[i]     = p0[i];
    a[i + 8] = p1[i];
  }
  return a;
}

// B fragment (32x16 f16, KxN): lane col = lane&15, hs = lane>>4.
// element e: K = (e>>3)*16 + hs*8 + (e&7). colk points at column n's K-major data (+kbase).
__device__ __forceinline__ v16h load_b_f16v(const _Float16* colk, int hs) {
  v8h_t lo = *(const v8h_t*)(colk + hs * 8);
  v8h_t hi = *(const v8h_t*)(colk + 16 + hs * 8);
  v16h b;
#pragma unroll
  for (int e = 0; e < 8; ++e) {
    b[e]     = lo[e];
    b[e + 8] = hi[e];
  }
  return b;
}

// ---------------- Kernel 1a: generic fp32 -> f16 conversion ----------------
__global__ void k_cvt_f16(const float* __restrict__ src,
                          _Float16* __restrict__ dst, int n) {
  for (int i = blockIdx.x * blockDim.x + threadIdx.x; i < n;
       i += gridDim.x * blockDim.x)
    dst[i] = (_Float16)src[i];
}

// ---------------- Kernel 1b: fuse W_XBC + W_gate -> f16 [h][256][64] --------
__global__ void k_cvt_wp(const float* __restrict__ W_XBC,
                         const float* __restrict__ W_gate,
                         _Float16* __restrict__ Wp) {
  const int n = Hn * PR * CH;
  for (int i = blockIdx.x * blockDim.x + threadIdx.x; i < n;
       i += gridDim.x * blockDim.x) {
    const int c = i & (CH - 1);
    const int r = (i >> 6) & (PR - 1);
    const int h = i >> 14;  // / (PR*CH)
    const float v = (r < 192) ? W_XBC[((size_t)h * 192 + r) * CH + c]
                              : W_gate[((size_t)h * CH + (r - 192)) * CH + c];
    Wp[i] = (_Float16)v;
  }
}

// ---------------- Kernel 2: logA + cumulative sum per (b,h) ----------------
__global__ void k_logA_cum(const float* __restrict__ inp,
                           const float* __restrict__ W_logA,
                           const float* __restrict__ b_logA,
                           float* __restrict__ cum) {
  __shared__ float sl[Tn];
  const int bh = blockIdx.x;       // [0, B*H)
  const int h  = bh & (Hn - 1);
  const int b  = bh >> 4;
  const float* w  = W_logA + h * CH;
  const float  bA = b_logA[h];
  for (int t = threadIdx.x; t < Tn; t += blockDim.x) {
    const float* xp = inp + ((size_t)(b * Tn + t) * Dn + h * CH);
    float s = 0.f;
#pragma unroll
    for (int c = 0; c < CH; ++c) s += w[c] * xp[c];
    sl[t] = s + bA;
  }
  __syncthreads();
  if (threadIdx.x == 0) {
    float s = 0.f;
    float* cp = cum + (size_t)bh * Tn;
    for (int t = 0; t < Tn; ++t) {
      s += sl[t];
      cp[t] = s;
    }
  }
}

// ---------------- Kernel 3: fused XBC + gate projection (WMMA) ----------------
// Per wave: one 16(t) x 16(n) tile of the 256-wide fused projection
// (n<192 -> silu(XBC), n>=192 -> gate). K = 64 in two WMMA steps.
// All operands pre-converted f16 -> pure fragment loads + WMMA, no cvt VALU.
__global__ __launch_bounds__(256) void k_proj(
    const _Float16* __restrict__ xh, const _Float16* __restrict__ Wp,
    const float* __restrict__ b_XBC, const float* __restrict__ b_gate,
    _Float16* __restrict__ Xt, _Float16* __restrict__ Bm,
    _Float16* __restrict__ Cm, float* __restrict__ gate) {
  const int wave  = blockIdx.x * (blockDim.x >> 5) + (threadIdx.x >> 5);
  const int ntile = wave & 15;          // 16 tiles of 16 -> 256 outputs
  const int mtile = (wave >> 4) & 63;   // 64 t-tiles of 16
  const int bh    = wave >> 10;         // 32
  const int h = bh & (Hn - 1), b = bh >> 4;
  const int lane = threadIdx.x & 31;
  const int ln = lane & 15, hs = lane >> 4;
  const int t0 = mtile * 16;
  const int n  = ntile * 16 + ln;

  const _Float16* wrow = Wp + ((size_t)h * PR + n) * CH;
  const _Float16* arow = xh + ((size_t)(b * Tn + t0 + ln) * Dn + h * CH);

  v8f acc = zero8();
#pragma unroll
  for (int s = 0; s < 2; ++s) {
    v16h a  = load_a_f16v(arow + s * 32, hs);
    v16h bb = load_b_f16v(wrow + s * 32, hs);
    acc = wmma_f16(a, bb, acc);
  }

  const float bias = (n < 192) ? b_XBC[h * 192 + n] : b_gate[h * CH + (n - 192)];
  const size_t bhT = (size_t)bh * Tn;
#pragma unroll
  for (int v = 0; v < 8; ++v) {
    const int t = t0 + hs * 8 + v;
    float val = acc[v] + bias;
    if (n < 192) {
      val = val / (1.f + __expf(-val));  // silu
      if (n < 64)
        Xt[((size_t)bh * CH + n) * Tn + t] = (_Float16)val;      // transposed
      else if (n < 128)
        Bm[(bhT + t) * SN + (n - 64)] = (_Float16)val;
      else
        Cm[(bhT + t) * SN + (n - 128)] = (_Float16)val;
    } else {
      gate[((size_t)(b * Tn + t) * Hn + h) * CH + (n - 192)] = val;
    }
  }
}

// ---------------- Kernel 4: fused causal scan (flash-style) ----------------
// Per wave: (b,h, 32-row i-tile). Loop j0 <= i0 in 32-steps:
//   G = Cm_i . Bm_j^T (WMMA, K=64), S = exp(cum_i - cum_j) causal mask * G,
//   S -> LDS (f16), Y += S . X_j (WMMA, K=32). Gate in epilogue.
// itile remapped (wv*4 + block-in-bh) so each block's triangular work is
// balanced (120..144 j-blocks/block instead of 36..228).
__global__ __launch_bounds__(256) void k_scan(
    const _Float16* __restrict__ Cm, const _Float16* __restrict__ Bm,
    const _Float16* __restrict__ Xt, const float* __restrict__ cum,
    const float* __restrict__ gate, _Float16* __restrict__ Yg) {
  __shared__ _Float16 sS[8][32 * 34];  // per-wave 32x32 tile, stride 34
  const int wv   = threadIdx.x >> 5;
  const int lane = threadIdx.x & 31;
  const int ln = lane & 15, hs = lane >> 4;
  const int wave  = blockIdx.x * 8 + wv;   // [0, 1024)
  const int local = wave & 31;
  const int itile = ((local & 7) << 2) + (local >> 3);  // balanced mapping
  const int bh    = wave >> 5;
  const int h = bh & (Hn - 1), b = bh >> 4;
  const int i0 = itile * 32;

  const float* cumb = cum + (size_t)bh * Tn;
  const _Float16* Cmb = Cm + (size_t)bh * Tn * SN;
  const _Float16* Bmb = Bm + (size_t)bh * Tn * SN;
  const _Float16* Xtb = Xt + (size_t)bh * CH * Tn;
  _Float16* sw = &sS[wv][0];

  // loop-invariant: cum_i and Cm A-fragments for both 16-row halves
  float ci[2][8];
  v16h aC[2][2];
#pragma unroll
  for (int mi = 0; mi < 2; ++mi) {
#pragma unroll
    for (int v = 0; v < 8; ++v) ci[mi][v] = cumb[i0 + mi * 16 + hs * 8 + v];
#pragma unroll
    for (int s = 0; s < 2; ++s)
      aC[mi][s] = load_a_f16v(Cmb + (size_t)(i0 + mi * 16 + ln) * SN + s * 32, hs);
  }

  v8f y[2][4];
#pragma unroll
  for (int mi = 0; mi < 2; ++mi)
#pragma unroll
    for (int nc = 0; nc < 4; ++nc) y[mi][nc] = zero8();

  for (int j0 = 0; j0 <= i0; j0 += 32) {
    // prefetch next Bm j-tile (32 rows x 64 f16 = 4KB), lane-spread
    if (j0 + 32 <= i0)
      __builtin_prefetch(Bmb + (size_t)(j0 + 32 + lane) * SN, 0, 3);

    // --- G = Cm . Bm^T, decay, -> LDS ---
    v16h bB[2][2];
#pragma unroll
    for (int jn = 0; jn < 2; ++jn)
#pragma unroll
      for (int s = 0; s < 2; ++s)
        bB[jn][s] =
            load_b_f16v(Bmb + (size_t)(j0 + jn * 16 + ln) * SN + s * 32, hs);

#pragma unroll
    for (int mi = 0; mi < 2; ++mi) {
#pragma unroll
      for (int jn = 0; jn < 2; ++jn) {
        v8f g = zero8();
        g = wmma_f16(aC[mi][0], bB[jn][0], g);
        g = wmma_f16(aC[mi][1], bB[jn][1], g);
        const int j  = j0 + jn * 16 + ln;
        const float cj = cumb[j];
#pragma unroll
        for (int v = 0; v < 8; ++v) {
          const int i = i0 + mi * 16 + hs * 8 + v;
          const float d = (j <= i) ? __expf(ci[mi][v] - cj) : 0.f;
          sw[(mi * 16 + hs * 8 + v) * 34 + jn * 16 + ln] =
              (_Float16)(g[v] * d);
        }
      }
    }
    // intra-wave LDS RAW fence (cross-lane: C-layout writes -> A-frag reads)
    asm volatile("s_wait_dscnt 0" ::: "memory");

    // --- Y += S . X_j ---
#pragma unroll
    for (int mi = 0; mi < 2; ++mi) {
      v16h aS = load_a_f16s(sw + (mi * 16 + ln) * 34, hs);
#pragma unroll
      for (int nc = 0; nc < 4; ++nc) {
        v16h bX =
            load_b_f16v(Xtb + (size_t)(nc * 16 + ln) * Tn + j0, hs);
        y[mi][nc] = wmma_f16(aS, bX, y[mi][nc]);
      }
    }
  }

  // epilogue: gate * Y -> Yg (f16)
#pragma unroll
  for (int mi = 0; mi < 2; ++mi)
#pragma unroll
    for (int nc = 0; nc < 4; ++nc)
#pragma unroll
      for (int v = 0; v < 8; ++v) {
        const int t = i0 + mi * 16 + hs * 8 + v;
        const int c = nc * 16 + ln;
        const float gv = gate[((size_t)(b * Tn + t) * Hn + h) * CH + c];
        Yg[(size_t)(b * Tn + t) * Dn + h * CH + c] = (_Float16)(gv * y[mi][nc][v]);
      }
}

// ---------------- Kernel 5: out = Yg @ W_out^T + b_out (WMMA) ----------------
// Per wave: 16(m) x 64(n) tile, K = 1024 in 32 WMMA steps.
__global__ __launch_bounds__(256) void k_out(
    const _Float16* __restrict__ Yg, const _Float16* __restrict__ W16,
    const float* __restrict__ b_out, float* __restrict__ out) {
  const int wave   = blockIdx.x * 8 + (threadIdx.x >> 5);  // [0, 2048)
  const int nstrip = wave & 15;
  const int mtile  = wave >> 4;  // [0, 128)
  const int lane = threadIdx.x & 31;
  const int ln = lane & 15, hs = lane >> 4;
  const int m0 = mtile * 16;
  const int n0 = nstrip * 64;

  v8f acc[4];
#pragma unroll
  for (int nc = 0; nc < 4; ++nc) acc[nc] = zero8();

  const _Float16* arow = Yg + (size_t)(m0 + ln) * Dn;
  for (int ks = 0; ks < Dn / 32; ++ks) {
    if (ks + 1 < Dn / 32)
      __builtin_prefetch(arow + (ks + 1) * 32, 0, 3);  // next A k-tile
    v16h a = load_a_f16v(arow + ks * 32, hs);
#pragma unroll
    for (int nc = 0; nc < 4; ++nc) {
      v16h bb =
          load_b_f16v(W16 + (size_t)(n0 + nc * 16 + ln) * Dn + ks * 32, hs);
      acc[nc] = wmma_f16(a, bb, acc[nc]);
    }
  }

#pragma unroll
  for (int nc = 0; nc < 4; ++nc)
#pragma unroll
    for (int v = 0; v < 8; ++v) {
      const int m  = m0 + hs * 8 + v;
      const int nn = n0 + nc * 16 + ln;
      out[(size_t)m * Dn + nn] = acc[nc][v] + b_out[nn];
    }
}

// ---------------- Host launch ----------------
extern "C" void kernel_launch(void* const* d_in, const int* in_sizes, int n_in,
                              void* d_out, int out_size, void* d_ws,
                              size_t ws_size, hipStream_t stream) {
  (void)in_sizes; (void)n_in; (void)out_size; (void)ws_size;
  const float* inp    = (const float*)d_in[0];
  const float* W_logA = (const float*)d_in[1];
  const float* b_logA = (const float*)d_in[2];
  const float* W_XBC  = (const float*)d_in[3];
  const float* b_XBC  = (const float*)d_in[4];
  const float* W_gate = (const float*)d_in[5];
  const float* b_gate = (const float*)d_in[6];
  const float* W_out  = (const float*)d_in[7];
  const float* b_out  = (const float*)d_in[8];
  float* out = (float*)d_out;

  // workspace carve-up (256B aligned)
  size_t off = 0;
  auto carve = [&](size_t bytes) {
    size_t o = off;
    off += (bytes + 255) & ~(size_t)255;
    return o;
  };
  char* ws = (char*)d_ws;
  float*    cum  = (float*)   (ws + carve((size_t)Bn * Hn * Tn * 4));        // 128 KB
  float*    gate = (float*)   (ws + carve((size_t)Bn * Tn * Hn * CH * 4));   // 8 MB
  _Float16* Xt   = (_Float16*)(ws + carve((size_t)Bn * Hn * CH * Tn * 2));   // 4 MB
  _Float16* Bm   = (_Float16*)(ws + carve((size_t)Bn * Hn * Tn * SN * 2));   // 4 MB
  _Float16* Cm   = (_Float16*)(ws + carve((size_t)Bn * Hn * Tn * SN * 2));   // 4 MB
  _Float16* Yg   = (_Float16*)(ws + carve((size_t)Bn * Tn * Dn * 2));        // 4 MB
  _Float16* W16  = (_Float16*)(ws + carve((size_t)Dn * Dn * 2));             // 2 MB
  _Float16* xh   = (_Float16*)(ws + carve((size_t)Bn * Tn * Dn * 2));        // 4 MB
  _Float16* Wp   = (_Float16*)(ws + carve((size_t)Hn * PR * CH * 2));        // 512 KB

  // 1) one-time f16 conversions (removes per-wave redundant cvt VALU work)
  k_cvt_f16<<<512, 256, 0, stream>>>(W_out, W16, Dn * Dn);
  k_cvt_f16<<<1024, 256, 0, stream>>>(inp, xh, Bn * Tn * Dn);
  k_cvt_wp<<<256, 256, 0, stream>>>(W_XBC, W_gate, Wp);
  // 2) logA + cumsum (fp32 for accuracy)
  k_logA_cum<<<Bn * Hn, 256, 0, stream>>>(inp, W_logA, b_logA, cum);
  // 3) fused projection: 32 bh * 64 m-tiles * 16 n-tiles = 32768 waves / 8
  k_proj<<<4096, 256, 0, stream>>>(xh, Wp, b_XBC, b_gate, Xt, Bm, Cm, gate);
  // 4) fused causal scan: 32 bh * 32 i-tiles = 1024 waves / 8
  k_scan<<<128, 256, 0, stream>>>(Cm, Bm, Xt, cum, gate, Yg);
  // 5) output GEMM: 128 m-tiles * 16 n-strips = 2048 waves / 8
  k_out<<<256, 256, 0, stream>>>(Yg, W16, b_out, out);
}
